// ConvGRUTSNet_57629871178053
// MI455X (gfx1250) — matrix operand order repaired and checked by reading, not compiled, not run on hardware
//
#include <hip/hip_runtime.h>

// ---------------------------------------------------------------------------
// Types for CDNA5 WMMA (wave32, 16x16x32 bf16 -> f32)
// ---------------------------------------------------------------------------
typedef __attribute__((ext_vector_type(16))) __bf16       v16bf;
typedef __attribute__((ext_vector_type(8)))  float        v8f;
typedef __attribute__((ext_vector_type(4)))  unsigned int v4u;

union FragU { v16bf v; v4u q[2]; };

// A-matrix 16x32 bf16 fragment (per ISA: lane<16 holds K {0..7,16..23},
// lane>=16 holds K {8..15,24..31}); caller passes row pointer already offset
// by ((lane&16)?8:0).
__device__ __forceinline__ v16bf ldA(const unsigned short* p) {
    FragU f;
    f.q[0] = *(const v4u*)(p);
    f.q[1] = *(const v4u*)(p + 16);
    return f.v;
}
// B-matrix 32x16 bf16 fragment: lane<16 holds K {0..7,16..23}, lane>=16
// holds K {8..15,24..31}; caller passes row pointer already offset by ((lane&16)?8:0).
__device__ __forceinline__ v16bf ldB(const unsigned short* p) {
    FragU f;
    f.q[0] = *(const v4u*)(p);
    f.q[1] = *(const v4u*)(p + 16);
    return f.v;
}

#define WMMA_BF16(a, b, c) \
    __builtin_amdgcn_wmma_f32_16x16x32_bf16(false, (a), false, (b), (short)0, (c), false, false)

__device__ __forceinline__ unsigned short f2bf(float f) {
    union { float f; unsigned int u; } v; v.f = f;
    unsigned int u = v.u;
    return (unsigned short)((u + 0x7FFFu + ((u >> 16) & 1u)) >> 16);
}
__device__ __forceinline__ float sigm(float x) { return 1.f / (1.f + __expf(-x)); }

#define Z8 {0.f,0.f,0.f,0.f,0.f,0.f,0.f,0.f}

// ---------------------------------------------------------------------------
// Weight convert + pad to bf16
// ---------------------------------------------------------------------------
__global__ void k_cvt_pad(const float* __restrict__ src, unsigned short* __restrict__ dst,
                          int srows, int scols, int drows, int dcols) {
    int i = blockIdx.x * blockDim.x + threadIdx.x;
    int n = drows * dcols;
    if (i >= n) return;
    int r = i / dcols, c = i - r * dcols;
    float v = (r < srows && c < scols) ? src[r * scols + c] : 0.f;
    dst[i] = f2bf(v);
}

// ---------------------------------------------------------------------------
// Kernel 1: cx GEMM.  out[(b,t), o(192), p(24)] = Wx[192,963] * im2col(x)
// grid = 512*7 blocks, 256 threads (8 waves).  K padded 963 -> 992.
// x slice (24 rows x 321 f32, contiguous) is staged into LDS with the CDNA5
// async global->LDS engine (ASYNCcnt) when available, then im2col'd to bf16.
// ---------------------------------------------------------------------------
#define CKP 992
__global__ __launch_bounds__(256) void k_cx(const float* __restrict__ x,
                                            const unsigned short* __restrict__ Wx,
                                            const float* __restrict__ zb,
                                            const float* __restrict__ rb,
                                            const float* __restrict__ hb,
                                            float* __restrict__ cxo) {
    __shared__ __align__(16) float          Xs[24 * 321];
    __shared__ __align__(16) unsigned short Bm[24][CKP];
    const int tid = threadIdx.x;
    const int b = blockIdx.x / 7, t = blockIdx.x % 7;
    const float* xb = x + ((size_t)b * 168 + (size_t)t * 24) * 321;

    // ---- stage x slice into LDS (async DMA path if the toolchain has it) ----
#if __has_builtin(__builtin_amdgcn_global_load_async_to_lds_b32)
    {
        typedef __attribute__((address_space(1))) int* gas1_i32;
        typedef __attribute__((address_space(3))) int* las3_i32;
        for (int i = tid; i < 24 * 321; i += 256) {
            __builtin_amdgcn_global_load_async_to_lds_b32(
                (gas1_i32)(xb + i), (las3_i32)(Xs + i), 0, 0);
        }
#if __has_builtin(__builtin_amdgcn_s_wait_asynccnt)
        __builtin_amdgcn_s_wait_asynccnt(0);
#else
        asm volatile("s_wait_asynccnt 0x0" ::: "memory");
#endif
    }
#else
    for (int i = tid; i < 24 * 321; i += 256) Xs[i] = xb[i];
#endif
    __syncthreads();

    // ---- build bf16 im2col tile in LDS ----
    for (int idx = tid; idx < 24 * CKP; idx += 256) {
        int p = idx / CKP, q = idx - p * CKP;
        float v = 0.f;
        if (q < 963) {
            int c = q / 3, k = q - c * 3;
            int j = p + k - 2;                  // left zero-pad of 2
            if (j >= 0) v = Xs[j * 321 + c];
        }
        Bm[p][q] = f2bf(v);
    }
    __syncthreads();

    const int wave = tid >> 5, lane = tid & 31, l15 = lane & 15;
    const int mb = (lane & 16) ? 8 : 0;
    for (int pair = wave; pair < 24; pair += 8) {
        int mt = pair % 12, nt = pair / 12;
        v8f acc = Z8;
        const unsigned short* Arow = Wx + (size_t)(mt * 16 + l15) * CKP + ((lane & 16) ? 8 : 0);
        int brow = nt * 16 + l15; if (brow > 23) brow = 23;
        const unsigned short* Brow = &Bm[brow][0] + ((lane & 16) ? 8 : 0);
        for (int ks = 0; ks < 31; ++ks) {
            int k0 = ks * 32;
            if (ks < 30) __builtin_prefetch(Arow + k0 + 32, 0, 1);
            v16bf af = ldA(Arow + k0);
            v16bf bf = ldB(Brow + k0);
            acc = WMMA_BF16(af, bf, acc);
        }
        int p = nt * 16 + l15;
        if (p < 24) {
            float* dst = cxo + ((size_t)b * 7 + t) * 192 * 24;
            #pragma unroll
            for (int r = 0; r < 8; ++r) {
                int o = mt * 16 + mb + r;
                float bias = (o < 64) ? zb[o] : ((o < 128) ? rb[o - 64] : hb[o - 128]);
                dst[o * 24 + p] = acc[r] + bias;
            }
        }
    }
}

// ---------------------------------------------------------------------------
// Kernel 2: ConvGRU recurrence over T=7 (per-b block) + gx = wih @ h GEMM.
// ---------------------------------------------------------------------------
__global__ __launch_bounds__(256) void k_convgru(const float* __restrict__ cx,
                                                 const unsigned short* __restrict__ Wh,
                                                 const float* __restrict__ zub,
                                                 const float* __restrict__ rub,
                                                 const float* __restrict__ hub,
                                                 const unsigned short* __restrict__ Wih,
                                                 const float* __restrict__ bih,
                                                 float* __restrict__ gxo) {
    __shared__ __align__(16) float          hst[64][24];
    __shared__ __align__(16) float          chout[192][24];
    __shared__ __align__(16) unsigned short Bh[24][192];
    __shared__ __align__(16) unsigned short Bg[24][64];
    const int tid = threadIdx.x, b = blockIdx.x;
    const int wave = tid >> 5, lane = tid & 31, l15 = lane & 15;
    const int mb = (lane & 16) ? 8 : 0;
    const float* cxb = cx + (size_t)b * 7 * 192 * 24;

    // t = 0: h = (1 - sigmoid(zx)) * tanh(hx)
    for (int idx = tid; idx < 64 * 24; idx += 256) {
        int o = idx / 24, p = idx - o * 24;
        float z = sigm(cxb[o * 24 + p]);
        float n = tanhf(cxb[(128 + o) * 24 + p]);
        hst[o][p] = (1.f - z) * n;
    }
    __syncthreads();

    for (int t = 1; t < 7; ++t) {
        const float* cxt = cxb + (size_t)t * 192 * 24;
        // im2col of padded h: K = 64*3 = 192
        for (int idx = tid; idx < 24 * 192; idx += 256) {
            int p = idx / 192, q = idx - p * 192;
            int c = q / 3, k = q - c * 3, j = p + k - 2;
            Bh[p][q] = f2bf(j >= 0 ? hst[c][j] : 0.f);
        }
        __syncthreads();
        for (int pair = wave; pair < 24; pair += 8) {
            int mt = pair % 12, nt = pair / 12;
            v8f acc = Z8;
            const unsigned short* Arow = Wh + (size_t)(mt * 16 + l15) * 192 + ((lane & 16) ? 8 : 0);
            int brow = nt * 16 + l15; if (brow > 23) brow = 23;
            const unsigned short* Brow = &Bh[brow][0] + ((lane & 16) ? 8 : 0);
            #pragma unroll
            for (int ks = 0; ks < 6; ++ks) {
                v16bf af = ldA(Arow + ks * 32);
                v16bf bf = ldB(Brow + ks * 32);
                acc = WMMA_BF16(af, bf, acc);
            }
            int p = nt * 16 + l15;
            if (p < 24) {
                #pragma unroll
                for (int r = 0; r < 8; ++r) {
                    int o = mt * 16 + mb + r;
                    float bias = (o < 64) ? zub[o] : ((o < 128) ? rub[o - 64] : hub[o - 128]);
                    chout[o][p] = acc[r] + bias;
                }
            }
        }
        __syncthreads();
        for (int idx = tid; idx < 64 * 24; idx += 256) {
            int o = idx / 24, p = idx - o * 24;
            float z = sigm(cxt[o * 24 + p]        + chout[o][p]);
            float r = sigm(cxt[(64 + o) * 24 + p] + chout[64 + o][p]);
            float n = tanhf(cxt[(128 + o) * 24 + p] + r * chout[128 + o][p]);
            hst[o][p] = (1.f - z) * n + z * hst[o][p];
        }
        __syncthreads();
    }

    // gx[b,p,g] = sum_o h[o][p] * wih[g][o] + bih[g]
    for (int idx = tid; idx < 24 * 64; idx += 256) {
        int p = idx / 64, o = idx - p * 64;
        Bg[p][o] = f2bf(hst[o][p]);
    }
    __syncthreads();
    for (int pair = wave; pair < 48; pair += 8) {
        int mt = pair % 24, nt = pair / 24;
        v8f acc = Z8;
        const unsigned short* Arow = Wih + (size_t)(mt * 16 + l15) * 64 + ((lane & 16) ? 8 : 0);
        int brow = nt * 16 + l15; if (brow > 23) brow = 23;
        const unsigned short* Brow = &Bg[brow][0] + ((lane & 16) ? 8 : 0);
        #pragma unroll
        for (int ks = 0; ks < 2; ++ks) {
            v16bf af = ldA(Arow + ks * 32);
            v16bf bf = ldB(Brow + ks * 32);
            acc = WMMA_BF16(af, bf, acc);
        }
        int p = nt * 16 + l15;
        if (p < 24) {
            #pragma unroll
            for (int r = 0; r < 8; ++r) {
                int g = mt * 16 + mb + r;
                gxo[((size_t)b * 24 + p) * 384 + g] = acc[r] + bih[g];
            }
        }
    }
}

// ---------------------------------------------------------------------------
// Kernel 3: GRU over P=24 steps, H=128.  16 batch rows per block, grid=32.
// Per step: gh[16,384] = hprev[16,128] @ whh[384,128]^T via WMMA.
// ---------------------------------------------------------------------------
__global__ __launch_bounds__(256) void k_gru(const float* __restrict__ gx,
                                             const unsigned short* __restrict__ Whh,
                                             const float* __restrict__ bhh,
                                             float* __restrict__ y) {
    __shared__ __align__(16) float          hprev[16][128];
    __shared__ __align__(16) unsigned short hb16[16][128];
    __shared__ __align__(16) float          gh[16][384];
    const int tid = threadIdx.x, b0 = blockIdx.x * 16;
    const int wave = tid >> 5, lane = tid & 31, l15 = lane & 15;
    const int mb = (lane & 16) ? 8 : 0;

    for (int idx = tid; idx < 16 * 128; idx += 256) ((float*)hprev)[idx] = 0.f;
    __syncthreads();

    for (int p = 0; p < 24; ++p) {
        for (int idx = tid; idx < 16 * 128; idx += 256)
            ((unsigned short*)hb16)[idx] = f2bf(((const float*)hprev)[idx]);
        __syncthreads();
        for (int pair = wave; pair < 24; pair += 8) {
            int nt = pair;
            v8f acc = Z8;
            const unsigned short* Arow = &hb16[l15][0] + ((lane & 16) ? 8 : 0);
            const unsigned short* Brow = Whh + (size_t)(nt * 16 + l15) * 128 + ((lane & 16) ? 8 : 0);
            #pragma unroll
            for (int ks = 0; ks < 4; ++ks) {
                v16bf af = ldA(Arow + ks * 32);
                v16bf bf = ldB(Brow + ks * 32);
                acc = WMMA_BF16(af, bf, acc);
            }
            int g = nt * 16 + l15;
            #pragma unroll
            for (int r = 0; r < 8; ++r) gh[mb + r][g] = acc[r];
        }
        __syncthreads();
        for (int idx = tid; idx < 16 * 128; idx += 256) {
            int bl = idx >> 7, hI = idx & 127;
            const float* gxv = gx + ((size_t)(b0 + bl) * 24 + p) * 384;
            float r  = sigm(gxv[hI]       + gh[bl][hI]       + bhh[hI]);
            float z  = sigm(gxv[128 + hI] + gh[bl][128 + hI] + bhh[128 + hI]);
            float n  = tanhf(gxv[256 + hI] + r * (gh[bl][256 + hI] + bhh[256 + hI]));
            float hn = (1.f - z) * n + z * hprev[bl][hI];
            hprev[bl][hI] = hn;
            y[((size_t)(b0 + bl) * 24 + p) * 128 + hI] = hn;
        }
        __syncthreads();
    }
}

// ---------------------------------------------------------------------------
// Kernel 4: tf attention branch (per-b block).
// ---------------------------------------------------------------------------
__global__ __launch_bounds__(256) void k_tf(const float* __restrict__ y,
                                            const unsigned short* __restrict__ Wc,
                                            const float* __restrict__ cb,
                                            const unsigned short* __restrict__ Wm1,
                                            const float* __restrict__ m1b,
                                            const float* __restrict__ m2w,
                                            const float* __restrict__ m2b,
                                            float* __restrict__ tfatt) {
    __shared__ __align__(16) unsigned short Bc[24][384];
    __shared__ __align__(16) unsigned short Bm1[24][256];
    __shared__ __align__(16) float          at1[24][128];
    __shared__ float at2[24];
    const int tid = threadIdx.x, b = blockIdx.x;
    const int wave = tid >> 5, lane = tid & 31, l15 = lane & 15;
    const int mb = (lane & 16) ? 8 : 0;
    const float* yb = y + (size_t)b * 24 * 128;
    const float* qv = yb + 23 * 128;

    for (int idx = tid; idx < 24 * 384; idx += 256) {
        int p = idx / 384, q = idx - p * 384;
        float v = 0.f;
        if (p < 23) {
            int ci = q / 3, k = q - ci * 3, j = p + k - 1;
            if (j >= 0 && j < 23) v = yb[j * 128 + ci];
        }
        Bc[p][q] = f2bf(v);
    }
    for (int idx = tid; idx < 24 * 128; idx += 256) {
        int p = idx / 128, i = idx - p * 128;
        Bm1[p][128 + i] = f2bf(p < 23 ? qv[i] : 0.f);
        if (p == 23) Bm1[23][i] = 0;
    }
    __syncthreads();
    // conv GEMM: M=128, K=384, N=23
    for (int pair = wave; pair < 16; pair += 8) {
        int mt = pair & 7, nt = pair >> 3;
        v8f acc = Z8;
        const unsigned short* Arow = Wc + (size_t)(mt * 16 + l15) * 384 + ((lane & 16) ? 8 : 0);
        int brow = nt * 16 + l15; if (brow > 23) brow = 23;
        const unsigned short* Brow = &Bc[brow][0] + ((lane & 16) ? 8 : 0);
        #pragma unroll
        for (int ks = 0; ks < 12; ++ks) {
            v16bf af = ldA(Arow + ks * 32);
            v16bf bf = ldB(Brow + ks * 32);
            acc = WMMA_BF16(af, bf, acc);
        }
        int p = nt * 16 + l15;
        if (p < 23) {
            #pragma unroll
            for (int r = 0; r < 8; ++r) {
                int co = mt * 16 + mb + r;
                Bm1[p][co] = f2bf(tanhf(acc[r] + cb[co]));
            }
        }
    }
    __syncthreads();
    // m1 GEMM: M=128, K=256, N=23
    for (int pair = wave; pair < 16; pair += 8) {
        int mt = pair & 7, nt = pair >> 3;
        v8f acc = Z8;
        const unsigned short* Arow = Wm1 + (size_t)(mt * 16 + l15) * 256 + ((lane & 16) ? 8 : 0);
        int brow = nt * 16 + l15; if (brow > 23) brow = 23;
        const unsigned short* Brow = &Bm1[brow][0] + ((lane & 16) ? 8 : 0);
        #pragma unroll
        for (int ks = 0; ks < 8; ++ks) {
            v16bf af = ldA(Arow + ks * 32);
            v16bf bf = ldB(Brow + ks * 32);
            acc = WMMA_BF16(af, bf, acc);
        }
        int p = nt * 16 + l15;
        if (p < 23) {
            #pragma unroll
            for (int r = 0; r < 8; ++r) {
                int m = mt * 16 + mb + r;
                at1[p][m] = tanhf(acc[r] + m1b[m]);
            }
        }
    }
    __syncthreads();
    if (tid < 23) {
        float s = m2b[0];
        for (int m = 0; m < 128; ++m) s += at1[tid][m] * m2w[m];
        at2[tid] = tanhf(s);
    }
    __syncthreads();
    if (tid < 128) {
        float s = 0.f;
        for (int p = 0; p < 23; ++p) s += yb[p * 128 + tid] * at2[p];
        tfatt[(size_t)b * 128 + tid] = s * (1.f / 23.f);
    }
}

// ---------------------------------------------------------------------------
// Kernel 5: sf attention branch (per-b block).  m2 via ds_add_f32 atomics.
// ---------------------------------------------------------------------------
__global__ __launch_bounds__(256) void k_sf(const float* __restrict__ y,
                                            const unsigned short* __restrict__ Wc,   // [32][96]
                                            const float* __restrict__ cb,            // [23]
                                            const unsigned short* __restrict__ Wm1,  // [80][160]
                                            const float* __restrict__ m1b,           // [75]
                                            const float* __restrict__ m2w,
                                            const float* __restrict__ m2b,
                                            float* __restrict__ sfatt) {
    __shared__ __align__(16) unsigned short Bs[128][96];
    __shared__ __align__(16) unsigned short Bm1[128][160];
    __shared__ float as2acc[128];
    __shared__ float as2[128];
    const int tid = threadIdx.x, b = blockIdx.x;
    const int wave = tid >> 5, lane = tid & 31, l15 = lane & 15;
    const int mb = (lane & 16) ? 8 : 0;
    const float* yb = y + (size_t)b * 24 * 128;
    const float* qv = yb + 23 * 128;

    for (int idx = tid; idx < 128 * 96; idx += 256) {
        int pos = idx / 96, q = idx - pos * 96;
        float v = 0.f;
        if (q < 69) {
            int pi = q / 3, k = q - pi * 3, j = pos + k - 1;
            if (j >= 0 && j < 128) v = yb[pi * 128 + j];
        }
        Bs[pos][q] = f2bf(v);
    }
    for (int idx = tid; idx < 128 * 160; idx += 256) {
        int c2 = idx / 160, q = idx - c2 * 160;
        if (q >= 23) Bm1[c2][q] = f2bf(q < 151 ? qv[q - 23] : 0.f);
    }
    if (tid < 128) as2acc[tid] = 0.f;
    __syncthreads();
    // conv GEMM: M=23(pad32), K=69(pad96), N=128
    for (int pair = wave; pair < 16; pair += 8) {
        int mt = pair & 1, nt = pair >> 1;
        v8f acc = Z8;
        const unsigned short* Arow = Wc + (size_t)(mt * 16 + l15) * 96 + ((lane & 16) ? 8 : 0);
        const unsigned short* Brow = &Bs[nt * 16 + l15][0] + ((lane & 16) ? 8 : 0);
        #pragma unroll
        for (int ks = 0; ks < 3; ++ks) {
            v16bf af = ldA(Arow + ks * 32);
            v16bf bf = ldB(Brow + ks * 32);
            acc = WMMA_BF16(af, bf, acc);
        }
        int c2 = nt * 16 + l15;
        #pragma unroll
        for (int r = 0; r < 8; ++r) {
            int po = mt * 16 + mb + r;
            if (po < 23) Bm1[c2][po] = f2bf(tanhf(acc[r] + cb[po]));
        }
    }
    __syncthreads();
    // m1 GEMM: M=75(pad80), K=151(pad160), N=128 ; fused m2 partial reduction
    for (int pair = wave; pair < 40; pair += 8) {
        int mt = pair % 5, nt = pair / 5;
        v8f acc = Z8;
        const unsigned short* Arow = Wm1 + (size_t)(mt * 16 + l15) * 160 + ((lane & 16) ? 8 : 0);
        const unsigned short* Brow = &Bm1[nt * 16 + l15][0] + ((lane & 16) ? 8 : 0);
        #pragma unroll
        for (int ks = 0; ks < 5; ++ks) {
            v16bf af = ldA(Arow + ks * 32);
            v16bf bf = ldB(Brow + ks * 32);
            acc = WMMA_BF16(af, bf, acc);
        }
        int c2 = nt * 16 + l15;
        float part = 0.f;
        #pragma unroll
        for (int r = 0; r < 8; ++r) {
            int m = mt * 16 + mb + r;
            if (m < 75) part += tanhf(acc[r] + m1b[m]) * m2w[m];
        }
        atomicAdd(&as2acc[c2], part);   // ds_add_f32
    }
    __syncthreads();
    if (tid < 128) as2[tid] = tanhf(as2acc[tid] + m2b[0]);
    __syncthreads();
    if (tid < 23) {
        float s = 0.f;
        for (int c2 = 0; c2 < 128; ++c2) s += yb[tid * 128 + c2] * as2[c2];
        sfatt[(size_t)b * 23 + tid] = s * (1.f / 128.f);
    }
}

// ---------------------------------------------------------------------------
// Kernel 6: final fc + autoregressive tail
// ---------------------------------------------------------------------------
__global__ __launch_bounds__(352) void k_final(const float* __restrict__ x,
                                               const float* __restrict__ tfatt,
                                               const float* __restrict__ sfatt,
                                               const float* __restrict__ y,
                                               const float* __restrict__ fcw,
                                               const float* __restrict__ fcb,
                                               const float* __restrict__ arw,
                                               const float* __restrict__ arb,
                                               float* __restrict__ out) {
    __shared__ float att[279];
    __shared__ float arws[24];
    const int tid = threadIdx.x, b = blockIdx.x;
    if (tid < 128)       att[tid] = tfatt[(size_t)b * 128 + tid];
    else if (tid < 151)  att[tid] = sfatt[(size_t)b * 23 + (tid - 128)];
    else if (tid < 279)  att[tid] = y[((size_t)b * 24 + 23) * 128 + (tid - 151)];
    if (tid < 24) arws[tid] = arw[tid];
    __syncthreads();
    if (tid < 321) {
        float s = fcb[tid];
        const float* w = fcw + (size_t)tid * 279;
        for (int j = 0; j < 279; ++j) s += att[j] * w[j];
        float lb = arb[0];
        const float* xr = x + ((size_t)b * 168 + 144) * 321 + tid;
        for (int p = 0; p < 24; ++p) lb += xr[(size_t)p * 321] * arws[p];
        out[(size_t)b * 321 + tid] = s + lb;
    }
}

// ---------------------------------------------------------------------------
// Host launcher
// ---------------------------------------------------------------------------
extern "C" void kernel_launch(void* const* d_in, const int* in_sizes, int n_in,
                              void* d_out, int out_size, void* d_ws, size_t ws_size,
                              hipStream_t stream) {
    (void)in_sizes; (void)n_in; (void)out_size; (void)ws_size;
    const float* x        = (const float*)d_in[0];
    const float* zwx_w    = (const float*)d_in[1];
    const float* zwx_b    = (const float*)d_in[2];
    const float* rwx_w    = (const float*)d_in[3];
    const float* rwx_b    = (const float*)d_in[4];
    const float* hwx_w    = (const float*)d_in[5];
    const float* hwx_b    = (const float*)d_in[6];
    const float* zuh_w    = (const float*)d_in[7];
    const float* zuh_b    = (const float*)d_in[8];
    const float* ruh_w    = (const float*)d_in[9];
    const float* ruh_b    = (const float*)d_in[10];
    const float* huh_w    = (const float*)d_in[11];
    const float* huh_b    = (const float*)d_in[12];
    const float* gru_wih  = (const float*)d_in[13];
    const float* gru_whh  = (const float*)d_in[14];
    const float* gru_bih  = (const float*)d_in[15];
    const float* gru_bhh  = (const float*)d_in[16];
    const float* tf_conv_w = (const float*)d_in[17];
    const float* tf_conv_b = (const float*)d_in[18];
    const float* tf_m1_w   = (const float*)d_in[19];
    const float* tf_m1_b   = (const float*)d_in[20];
    const float* tf_m2_w   = (const float*)d_in[21];
    const float* tf_m2_b   = (const float*)d_in[22];
    const float* sf_conv_w = (const float*)d_in[23];
    const float* sf_conv_b = (const float*)d_in[24];
    const float* sf_m1_w   = (const float*)d_in[25];
    const float* sf_m1_b   = (const float*)d_in[26];
    const float* sf_m2_w   = (const float*)d_in[27];
    const float* sf_m2_b   = (const float*)d_in[28];
    const float* fc_w      = (const float*)d_in[29];
    const float* fc_b      = (const float*)d_in[30];
    const float* ar_w      = (const float*)d_in[31];
    const float* ar_b      = (const float*)d_in[32];

    char* ws = (char*)d_ws;
    size_t off = 0;
    auto take = [&](size_t bytes) -> char* {
        char* p = ws + off;
        off = (off + bytes + 255) & ~(size_t)255;
        return p;
    };
    unsigned short* Wx    = (unsigned short*)take((size_t)192 * 992 * 2);
    unsigned short* Wh    = (unsigned short*)take((size_t)192 * 192 * 2);
    unsigned short* Wih   = (unsigned short*)take((size_t)384 * 64 * 2);
    unsigned short* Whh   = (unsigned short*)take((size_t)384 * 128 * 2);
    unsigned short* Wtfc  = (unsigned short*)take((size_t)128 * 384 * 2);
    unsigned short* Wtfm1 = (unsigned short*)take((size_t)128 * 256 * 2);
    unsigned short* Wsfc  = (unsigned short*)take((size_t)32 * 96 * 2);
    unsigned short* Wsfm1 = (unsigned short*)take((size_t)80 * 160 * 2);
    float* cxb   = (float*)take((size_t)512 * 7 * 192 * 24 * 4);
    float* gxb   = (float*)take((size_t)512 * 24 * 384 * 4);
    float* yb    = (float*)take((size_t)512 * 24 * 128 * 4);
    float* tfatt = (float*)take((size_t)512 * 128 * 4);
    float* sfatt = (float*)take((size_t)512 * 23 * 4);

    auto cvt = [&](const float* src, unsigned short* dst, int sr, int sc, int dr, int dc) {
        int n = dr * dc;
        k_cvt_pad<<<(n + 255) / 256, 256, 0, stream>>>(src, dst, sr, sc, dr, dc);
    };
    cvt(zwx_w, Wx,             64, 963, 64, 992);
    cvt(rwx_w, Wx + 64 * 992,  64, 963, 64, 992);
    cvt(hwx_w, Wx + 128 * 992, 64, 963, 64, 992);
    cvt(zuh_w, Wh,             64, 192, 64, 192);
    cvt(ruh_w, Wh + 64 * 192,  64, 192, 64, 192);
    cvt(huh_w, Wh + 128 * 192, 64, 192, 64, 192);
    cvt(gru_wih,  Wih,   384, 64, 384, 64);
    cvt(gru_whh,  Whh,   384, 128, 384, 128);
    cvt(tf_conv_w, Wtfc, 128, 384, 128, 384);
    cvt(tf_m1_w,  Wtfm1, 128, 256, 128, 256);
    cvt(sf_conv_w, Wsfc, 23, 69, 32, 96);
    cvt(sf_m1_w,  Wsfm1, 75, 151, 80, 160);

    k_cx<<<512 * 7, 256, 0, stream>>>(x, Wx, zwx_b, rwx_b, hwx_b, cxb);
    k_convgru<<<512, 256, 0, stream>>>(cxb, Wh, zuh_b, ruh_b, huh_b, Wih, gru_bih, gxb);
    k_gru<<<32, 256, 0, stream>>>(gxb, Whh, gru_bhh, yb);
    k_tf<<<512, 256, 0, stream>>>(yb, Wtfc, tf_conv_b, Wtfm1, tf_m1_b, tf_m2_w, tf_m2_b, tfatt);
    k_sf<<<512, 256, 0, stream>>>(yb, Wsfc, sf_conv_b, Wsfm1, sf_m1_b, sf_m2_w, sf_m2_b, sfatt);
    k_final<<<512, 352, 0, stream>>>(x, tfatt, sfatt, yb, fc_w, fc_b, ar_w, ar_b, (float*)d_out);
}
